// AutoCorrelation_1013612282368
// MI455X (gfx1250) — matrix operand order repaired and checked by reading, not compiled, not run on hardware
//
#include <hip/hip_runtime.h>
#include <hip/hip_bf16.h>
#include <stdint.h>

// ---------------------------------------------------------------------------
// Problem constants (from reference)
// ---------------------------------------------------------------------------
#define BATCH 16
#define SEQL  4096
#define EMB   512
#define NHEAD 8
#define HDIM  64          // EMB / NHEAD
#define KTOP  16          // int(2.0 * log(4096))
#define MROWS (BATCH * SEQL)   // 65536
#define NFREQ 2049        // SEQL/2 + 1 (Hermitian half-spectrum)

// ---------------------------------------------------------------------------
// WMMA types (CDNA5 / gfx1250, wave32)
// ---------------------------------------------------------------------------
typedef __attribute__((ext_vector_type(16))) __bf16 v16bf;
typedef __attribute__((ext_vector_type(8)))  float  v8f;

__device__ __forceinline__ uint16_t f2bf(float x) {
    // round-to-nearest-even float -> bf16
    uint32_t u = __float_as_uint(x);
    uint32_t r = u + 0x7FFFu + ((u >> 16) & 1u);
    return (uint16_t)(r >> 16);
}

// ---------------------------------------------------------------------------
// GEMM: Y[m,n] = sum_k X[m,k] * W[n,k] + bias[n]
//   X: (M,K) fp32 row-major, W: (N,K) fp32 row-major (nn.Linear weight).
//   transposed == 0 : Y stored (M,N) fp32 row-major.
//   transposed == 1 : Y stored (B, N, L) fp32 with m = b*L + t  (so the FFT
//                     stage reads each channel's time series contiguously).
// Block tile 128x128x32, 256 threads = 8 waves in a 4(M) x 2(N) grid,
// each wave computes a 32x64 tile = 2x4 fragments of 16x16.
// Double-buffered LDS: next K-tile staged via registers while WMMAs run.
// ---------------------------------------------------------------------------
#define BM 128
#define BN 128
#define BK 32
#define LDT (BK + 2)      // uint16 row stride in LDS (even => dword-aligned pairs)

union FragU { v16bf v; uint32_t u[8]; };

// A fragment: 16x32 bf16, rows = M.  ISA layout:
//   lanes 0-15: M=0..15; lanes 16-31: M=0..15 (second K half of each 16-block)
//   VGPR j<4 : K = 2j   + 8*half ; VGPR j>=4: K = 16+2(j-4) + 8*half
__device__ __forceinline__ v16bf load_frag_a(const uint16_t* base, int lane) {
    int r = lane & 15, half = lane >> 4;
    const uint16_t* p = base + r * LDT;
    FragU f;
#pragma unroll
    for (int j = 0; j < 8; ++j) {
        int kb = ((j < 4) ? (2 * j) : (16 + 2 * (j - 4))) + 8 * half;
        f.u[j] = *reinterpret_cast<const uint32_t*>(p + kb);
    }
    return f.v;
}

// B fragment: 32x16 bf16 (KxN), lane&15 = N, same K layout as A:
// VGPR j<4 : K = 2j + 8*half ; VGPR j>=4: K = 16+2(j-4) + 8*half.
// LDS tile is stored as Bs[n][k] so the K-pair is contiguous.
__device__ __forceinline__ v16bf load_frag_b(const uint16_t* base, int lane) {
    int r = lane & 15, half = lane >> 4;
    const uint16_t* p = base + r * LDT;
    FragU f;
#pragma unroll
    for (int j = 0; j < 8; ++j) {
        int kb = ((j < 4) ? (2 * j) : (16 + 2 * (j - 4))) + 8 * half;
        f.u[j] = *reinterpret_cast<const uint32_t*>(p + kb);
    }
    return f.v;
}

__global__ __launch_bounds__(256)
void gemm_xwT_bias(const float* __restrict__ X, const float* __restrict__ W,
                   const float* __restrict__ bias, float* __restrict__ Y,
                   int M, int N, int K, int transposed) {
    __shared__ uint16_t As[2][BM * LDT];
    __shared__ uint16_t Bs[2][BN * LDT];

    const int tid  = threadIdx.x;
    const int lane = tid & 31;
    const int wave = tid >> 5;
    const int wm = wave & 3;   // 4 waves along M: 4 * 32 = 128
    const int wn = wave >> 2;  // 2 waves along N: 2 * 64 = 128
    const int bm = blockIdx.x * BM;
    const int bn = blockIdx.y * BN;
    const int NT = K / BK;

    // per-thread staging registers: 4 float4 for A tile, 4 for B tile
    float4 ra[4], rb[4];
    const int r_ld  = tid >> 1;              // unused pattern kept simple below

    auto load_tile = [&](int kt) {
#pragma unroll
        for (int v = 0; v < 4; ++v) {
            int idx = v * 256 + tid;          // 0..1023 float4 slots
            int r   = idx >> 3;               // 8 float4 per 32-wide row
            int c4  = (idx & 7) << 2;
            ra[v] = *reinterpret_cast<const float4*>(
                X + (size_t)(bm + r) * K + kt + c4);
            rb[v] = *reinterpret_cast<const float4*>(
                W + (size_t)(bn + r) * K + kt + c4);
        }
    };
    auto store_tile = [&](int buf) {
#pragma unroll
        for (int v = 0; v < 4; ++v) {
            int idx = v * 256 + tid;
            int r   = idx >> 3;
            int c4  = (idx & 7) << 2;
            uint16_t* pa = &As[buf][r * LDT + c4];
            pa[0] = f2bf(ra[v].x); pa[1] = f2bf(ra[v].y);
            pa[2] = f2bf(ra[v].z); pa[3] = f2bf(ra[v].w);
            uint16_t* pb = &Bs[buf][r * LDT + c4];
            pb[0] = f2bf(rb[v].x); pb[1] = f2bf(rb[v].y);
            pb[2] = f2bf(rb[v].z); pb[3] = f2bf(rb[v].w);
        }
    };

    v8f acc[2][4];
#pragma unroll
    for (int i = 0; i < 2; ++i)
#pragma unroll
        for (int j = 0; j < 4; ++j)
            acc[i][j] = v8f{0.f, 0.f, 0.f, 0.f, 0.f, 0.f, 0.f, 0.f};

    (void)r_ld;
    load_tile(0);
    store_tile(0);
    __syncthreads();

    for (int it = 0; it < NT; ++it) {
        const int cur = it & 1;
        if (it + 1 < NT) {
            load_tile((it + 1) * BK);          // global loads in flight
            if (it + 2 < NT) {
                // gfx1250 speculative prefetch of the tile after next
                int idx = tid;
                int r   = idx >> 3;
                int c4  = (idx & 7) << 2;
                __builtin_prefetch(X + (size_t)(bm + r) * K + (it + 2) * BK + c4, 0, 1);
                __builtin_prefetch(W + (size_t)(bn + r) * K + (it + 2) * BK + c4, 0, 1);
            }
        }

        v16bf afrag[2], bfrag[4];
#pragma unroll
        for (int i = 0; i < 2; ++i)
            afrag[i] = load_frag_a(&As[cur][(wm * 32 + i * 16) * LDT], lane);
#pragma unroll
        for (int j = 0; j < 4; ++j)
            bfrag[j] = load_frag_b(&Bs[cur][(wn * 64 + j * 16) * LDT], lane);

#pragma unroll
        for (int i = 0; i < 2; ++i)
#pragma unroll
            for (int j = 0; j < 4; ++j)
                acc[i][j] = __builtin_amdgcn_wmma_f32_16x16x32_bf16(
                    false, afrag[i], false, bfrag[j],
                    (short)0, acc[i][j], false, false);

        if (it + 1 < NT) store_tile(cur ^ 1);  // fill other buffer
        __syncthreads();
    }

    // Epilogue: C/D layout: lane&15 = N; VGPR t: M = t + 8*(lane>>4).
    const int lr = lane & 15, half = lane >> 4;
    if (!transposed) {
#pragma unroll
        for (int j = 0; j < 4; ++j) {
            int n = bn + wn * 64 + j * 16 + lr;
            float bv = bias[n];
#pragma unroll
            for (int i = 0; i < 2; ++i) {
                int mbase = bm + wm * 32 + i * 16 + 8 * half;
#pragma unroll
                for (int t = 0; t < 8; ++t)
                    Y[(size_t)(mbase + t) * N + n] = acc[i][j][t] + bv;
            }
        }
    } else {
        // (B, N, L) layout: element (m, n) -> b = m / SEQL, t = m % SEQL
#pragma unroll
        for (int j = 0; j < 4; ++j) {
            int n = bn + wn * 64 + j * 16 + lr;
            float bv = bias[n];
#pragma unroll
            for (int i = 0; i < 2; ++i) {
                int mbase = bm + wm * 32 + i * 16 + 8 * half;
                int b     = mbase >> 12;          // / SEQL
                int tbase = mbase & (SEQL - 1);
                float* dst = Y + ((size_t)b * EMB + n) * SEQL + tbase;
#pragma unroll
                for (int t = 0; t < 8; ++t)
                    dst[t] = acc[i][j][t] + bv;   // 8 consecutive floats/lane
            }
        }
    }
}

// ---------------------------------------------------------------------------
// In-LDS radix-2 FFT, 4096 complex points, 256 threads.
// Caller must __syncthreads() after filling s[].
// ---------------------------------------------------------------------------
__device__ void fft4096(float2* s, int tid, bool inverse) {
    // bit reversal (12 bits)
    for (int i = tid; i < 4096; i += 256) {
        int j = (int)(__brev((unsigned)i) >> 20);
        if (j > i) { float2 t = s[i]; s[i] = s[j]; s[j] = t; }
    }
    __syncthreads();
    for (int len = 2; len <= 4096; len <<= 1) {
        int half = len >> 1;
        float ang = (inverse ? 6.28318530717958647f : -6.28318530717958647f)
                    / (float)len;
        for (int bfi = tid; bfi < 2048; bfi += 256) {
            int grp = bfi / half;          // power-of-two -> shift
            int pos = bfi & (half - 1);
            int i0 = grp * len + pos;
            int i1 = i0 + half;
            float sn, cs;
            __sincosf(ang * (float)pos, &sn, &cs);
            float2 u = s[i0], v = s[i1];
            float tr = v.x * cs - v.y * sn;
            float ti = v.x * sn + v.y * cs;
            s[i0] = make_float2(u.x + tr, u.y + ti);
            s[i1] = make_float2(u.x - tr, u.y - ti);
        }
        __syncthreads();
    }
}

// ---------------------------------------------------------------------------
// Per (b, e): FFT(q + i*k), split spectra, S = Qf * conj(Kf).
// Q/K are in (B, E, L) layout -> fully coalesced loads.
// Only the Hermitian half (w = 0..2048) is atomically accumulated.
// ---------------------------------------------------------------------------
__global__ __launch_bounds__(256)
void fft_xspec(const float* __restrict__ Qt, const float* __restrict__ Kt,
               float* __restrict__ accum) {
    __shared__ float2 z[SEQL];
    const int blk = blockIdx.x;
    const int b = blk >> 9;        // / 512
    const int e = blk & 511;
    const float* qb = Qt + ((size_t)b * EMB + e) * SEQL;
    const float* kb = Kt + ((size_t)b * EMB + e) * SEQL;
    for (int t = threadIdx.x; t < SEQL; t += 256)
        z[t] = make_float2(qb[t], kb[t]);
    __syncthreads();
    fft4096(z, threadIdx.x, false);

    float* acc = accum + (size_t)b * (2 * SEQL);
    for (int w = threadIdx.x; w < NFREQ; w += 256) {
        float2 Z  = z[w];
        float2 Zm = z[(SEQL - w) & (SEQL - 1)];
        // Qf = (Z + conj(Zm))/2 ; Kf = (Z - conj(Zm))/(2i)
        float qx = 0.5f * (Z.x + Zm.x), qy = 0.5f * (Z.y - Zm.y);
        float kx = 0.5f * (Z.y + Zm.y), ky = 0.5f * (Zm.x - Z.x);
        // S = Qf * conj(Kf)
        float sx = qx * kx + qy * ky;
        float sy = qy * kx - qx * ky;
        atomicAdd(&acc[2 * w + 0], sx);
        atomicAdd(&acc[2 * w + 1], sy);
    }
}

// ---------------------------------------------------------------------------
// Per batch: rebuild Hermitian spectrum, inverse FFT -> corr_mean (real part).
// Scale = 1/L (irfft) * 1/(H*D) (channel mean).
// ---------------------------------------------------------------------------
__global__ __launch_bounds__(256)
void ifft_corr(const float* __restrict__ accum, float* __restrict__ corr) {
    __shared__ float2 z[SEQL];
    const int b = blockIdx.x;
    const float* acc = accum + (size_t)b * (2 * SEQL);
    for (int w = threadIdx.x; w < NFREQ; w += 256) {
        float2 S = make_float2(acc[2 * w], acc[2 * w + 1]);
        z[w] = S;
        if (w > 0 && w < SEQL / 2)
            z[SEQL - w] = make_float2(S.x, -S.y);   // conj mirror
    }
    __syncthreads();
    fft4096(z, threadIdx.x, true);
    const float scale = 1.0f / ((float)SEQL * (float)EMB);
    for (int t = threadIdx.x; t < SEQL; t += 256)
        corr[(size_t)b * SEQL + t] = z[t].x * scale;
}

// ---------------------------------------------------------------------------
// Per batch: top-16 of corr_mean + softmax.
// ---------------------------------------------------------------------------
__global__ __launch_bounds__(256)
void topk_softmax(const float* __restrict__ corr,
                  float* __restrict__ probs, int* __restrict__ delays) {
    __shared__ float vals[SEQL];
    __shared__ float rv[256];
    __shared__ int   ri[256];
    __shared__ float topv[KTOP];
    __shared__ int   topi[KTOP];
    const int b = blockIdx.x;
    const int tid = threadIdx.x;
    for (int i = tid; i < SEQL; i += 256) vals[i] = corr[(size_t)b * SEQL + i];
    __syncthreads();
    for (int k = 0; k < KTOP; ++k) {
        float bv = -3.4e38f; int bi = 0;
        for (int i = tid; i < SEQL; i += 256)
            if (vals[i] > bv) { bv = vals[i]; bi = i; }
        rv[tid] = bv; ri[tid] = bi;
        __syncthreads();
        for (int s = 128; s > 0; s >>= 1) {
            if (tid < s) {
                float v2 = rv[tid + s]; int i2 = ri[tid + s];
                if (v2 > rv[tid] || (v2 == rv[tid] && i2 < ri[tid])) {
                    rv[tid] = v2; ri[tid] = i2;
                }
            }
            __syncthreads();
        }
        if (tid == 0) { topv[k] = rv[0]; topi[k] = ri[0]; vals[ri[0]] = -3.4e38f; }
        __syncthreads();
    }
    if (tid == 0) {
        float mx = topv[0];
        for (int k = 1; k < KTOP; ++k) mx = fmaxf(mx, topv[k]);
        float ex[KTOP], sum = 0.f;
        for (int k = 0; k < KTOP; ++k) { ex[k] = __expf(topv[k] - mx); sum += ex[k]; }
        float inv = 1.0f / sum;
        for (int k = 0; k < KTOP; ++k) {
            probs[b * KTOP + k]  = ex[k] * inv;
            delays[b * KTOP + k] = topi[k];
        }
    }
}

// ---------------------------------------------------------------------------
// ctx[b,l,e] = sum_i probs[b,i] * V[b,(l+delay_i) mod L, e]
// One block per (b,l); V rows replay out of the 192 MB L2 (V = 134 MB).
// ---------------------------------------------------------------------------
__global__ __launch_bounds__(256)
void aggregate(const float* __restrict__ V, const float* __restrict__ probs,
               const int* __restrict__ delays, float* __restrict__ ctx) {
    __shared__ float sp[KTOP];
    __shared__ int   sd[KTOP];
    const int bl = blockIdx.x;
    const int b = bl >> 12;            // / 4096
    const int l = bl & (SEQL - 1);
    if (threadIdx.x < KTOP) {
        sp[threadIdx.x] = probs[b * KTOP + threadIdx.x];
        sd[threadIdx.x] = delays[b * KTOP + threadIdx.x];
    }
    __syncthreads();
    const float* Vb = V + (size_t)b * SEQL * EMB;
    const int e0 = threadIdx.x, e1 = threadIdx.x + 256;
    float a0 = 0.f, a1 = 0.f;
#pragma unroll
    for (int i = 0; i < KTOP; ++i) {
        int src = (l + sd[i]) & (SEQL - 1);
        const float* row = Vb + (size_t)src * EMB;
        a0 += sp[i] * row[e0];
        a1 += sp[i] * row[e1];
    }
    float* out = ctx + (size_t)bl * EMB;
    out[e0] = a0; out[e1] = a1;
}

__global__ void zero_f32(float* __restrict__ p, int n) {
    int i = blockIdx.x * blockDim.x + threadIdx.x;
    if (i < n) p[i] = 0.f;
}

// ---------------------------------------------------------------------------
// Host launcher
// ---------------------------------------------------------------------------
extern "C" void kernel_launch(void* const* d_in, const int* in_sizes, int n_in,
                              void* d_out, int out_size, void* d_ws, size_t ws_size,
                              hipStream_t stream) {
    const float* queries = (const float*)d_in[0];
    const float* keys    = (const float*)d_in[1];
    const float* values  = (const float*)d_in[2];
    const float* Wq = (const float*)d_in[3];
    const float* bq = (const float*)d_in[4];
    const float* Wk = (const float*)d_in[5];
    const float* bk = (const float*)d_in[6];
    const float* Wv = (const float*)d_in[7];
    const float* bv = (const float*)d_in[8];
    const float* Wo = (const float*)d_in[9];
    const float* bo = (const float*)d_in[10];
    float* out = (float*)d_out;

    const size_t NP = (size_t)MROWS * EMB;   // 33,554,432 elems per tensor
    float* Qt    = (float*)d_ws;             // (B, E, L) transposed
    float* Kt    = Qt + NP;                  // (B, E, L) transposed
    float* Vp    = Kt + NP;                  // (B, L, E) row-major
    float* Ctx   = Vp + NP;                  // (B, L, E) row-major
    float* accum = Ctx + NP;                           // B * 2L floats
    float* corr  = accum + (size_t)BATCH * 2 * SEQL;   // B * L floats
    float* probs = corr + (size_t)BATCH * SEQL;        // B * KTOP floats
    int*   delays = (int*)(probs + BATCH * KTOP);      // B * KTOP ints

    dim3 gemm_grid(MROWS / BM, EMB / BN);   // (512, 4)

    // 1-3: projections (bf16 WMMA GEMMs); Q/K stored channel-major for the FFT
    gemm_xwT_bias<<<gemm_grid, 256, 0, stream>>>(queries, Wq, bq, Qt, MROWS, EMB, EMB, 1);
    gemm_xwT_bias<<<gemm_grid, 256, 0, stream>>>(keys,    Wk, bk, Kt, MROWS, EMB, EMB, 1);
    gemm_xwT_bias<<<gemm_grid, 256, 0, stream>>>(values,  Wv, bv, Vp, MROWS, EMB, EMB, 0);

    // 4: zero the spectrum accumulator (required each call: atomic accumulate)
    int acc_n = BATCH * 2 * SEQL;
    zero_f32<<<(acc_n + 255) / 256, 256, 0, stream>>>(accum, acc_n);

    // 5: per-(b,e) FFT + half-spectrum accumulation
    fft_xspec<<<BATCH * EMB, 256, 0, stream>>>(Qt, Kt, accum);

    // 6: per-batch inverse FFT -> corr_mean
    ifft_corr<<<BATCH, 256, 0, stream>>>(accum, corr);

    // 7: top-k + softmax
    topk_softmax<<<BATCH, 256, 0, stream>>>(corr, probs, delays);

    // 8: time-delay aggregation
    aggregate<<<BATCH * SEQL, 256, 0, stream>>>(Vp, probs, delays, Ctx);

    // 9: output projection into d_out
    gemm_xwT_bias<<<gemm_grid, 256, 0, stream>>>(Ctx, Wo, bo, out, MROWS, EMB, EMB, 0);
}
